// MapAgent_52819507806640
// MI455X (gfx1250) — matrix operand order repaired and checked by reading, not compiled, not run on hardware
//
#include <hip/hip_runtime.h>
#include <hip/hip_bf16.h>

typedef __attribute__((ext_vector_type(16))) _Float16 v16h;
typedef __attribute__((ext_vector_type(8)))  _Float16 v8h;
typedef __attribute__((ext_vector_type(4)))  _Float16 v4h;
typedef __attribute__((ext_vector_type(8)))  float    v8f;
typedef __attribute__((ext_vector_type(4)))  float    v4f;
typedef __attribute__((ext_vector_type(4)))  int      v4i;

constexpr int kT = 32, kB = 64, kTB = kT * kB;          // 2048
constexpr int kMAP = 8192;                              // 32*16*16
constexpr int kNACT = 5;

// ---------------------------------------------------------------------------
// WMMA fragment conventions (v_wmma_f32_16x16x32_f16):
//  A (16x32): lane&15 = row M; alo=(lane>>4)*8; two contiguous 8-half chunks.
//  B (32x16): pre-swizzled so each lane's 16 halfs are contiguous (32B).
//  C (16x16): lane&15 = col; reg r = row r + 8*(lane>>4).
// Swizzled B layout: ((nt*KC + kc)*32 + lane)*16 + j,
//                    k = kc*32 + (lane>>4)*8 + (j&7) + (j>>3)*16, n = nt*16 + (lane&15)
// ---------------------------------------------------------------------------
__device__ __forceinline__ v8f wmma(v16h a, v16h b, v8f c) {
  return __builtin_amdgcn_wmma_f32_16x16x32_f16(false, a, false, b,
                                                (short)0, c, false, false);
}
__device__ __forceinline__ v16h cat(v8h x, v8h y) {
  return __builtin_shufflevector(x, y, 0,1,2,3,4,5,6,7,8,9,10,11,12,13,14,15);
}
__device__ __forceinline__ void bsw_idx(int i, int KC, int& k, int& n) {
  int j = i & 15, lane = (i >> 4) & 31, rest = i >> 9;
  int kc = rest % KC, nt = rest / KC;
  k = kc * 32 + ((lane & 16) ? 8 : 0) + (j & 7) + (j >> 3) * 16;
  n = nt * 16 + (lane & 15);
}

// ---- optional CDNA5 async global->LDS staging (guarded; safe fallback) ----
#if __has_builtin(__builtin_amdgcn_global_load_async_to_lds_b128)
#define ASYNC_LDS 1
#endif
__device__ __forceinline__ void cp16_to_lds(const _Float16* g, _Float16* l) {
#ifdef ASYNC_LDS
  __builtin_amdgcn_global_load_async_to_lds_b128(
      (__attribute__((address_space(1))) v4i*)(const void*)g,
      (__attribute__((address_space(3))) v4i*)(void*)l, 0, 0);
#else
  *(v8h*)l = *(const v8h*)g;
#endif
}
__device__ __forceinline__ void async_wait() {
#ifdef ASYNC_LDS
#if __has_builtin(__builtin_amdgcn_s_wait_asynccnt)
  __builtin_amdgcn_s_wait_asynccnt(0);
#else
  asm volatile("s_wait_asynccnt 0" ::: "memory");
#endif
#endif
}

// ---- LDS-transposed epilogue: 16x32 f16 tile -> one 32B store per lane ---
__device__ __forceinline__ void epi_store32(_Float16* __restrict__ dst, int stride,
    int mbase, int ncol0, const v8f& acc0, const v8f& acc1,
    float bs0, float bs1, bool relu, _Float16* lds512, int lane) {
  const int hi = lane >> 4, col = lane & 15;
#pragma unroll
  for (int rI = 0; rI < 8; ++rI) {
    int row = rI + 8 * hi;
    float v0 = acc0[rI] + bs0, v1 = acc1[rI] + bs1;
    if (relu) { v0 = v0 > 0.f ? v0 : 0.f; v1 = v1 > 0.f ? v1 : 0.f; }
    lds512[row * 32 + col]      = (_Float16)v0;
    lds512[row * 32 + 16 + col] = (_Float16)v1;
  }
  asm volatile("" ::: "memory");   // LDS ops are wave-ordered; block reordering
  const int row = lane >> 1, seg = (lane & 1) * 16;
  v16h val = *(const v16h*)(lds512 + row * 32 + seg);
  *(v16h*)(dst + (size_t)(mbase + row) * stride + ncol0 + seg) = val;
}

// ---- B swizzle kernels (f32 weights -> f16 fragment-order) ---------------
__global__ void k_swz1(const float* __restrict__ w1, _Float16* __restrict__ d) {
  int i = blockIdx.x * 256 + threadIdx.x; if (i >= 2 * 6 * 512) return;
  int k, n; bsw_idx(i, 6, k, n);
  int c = k % 3, p = k / 3, ky = p >> 3, kx = p & 7;
  d[i] = (_Float16)w1[n * 192 + c * 64 + ky * 8 + kx];
}
__global__ void k_swz2(const float* __restrict__ w2, _Float16* __restrict__ d) {
  int i = blockIdx.x * 256 + threadIdx.x; if (i >= 4 * 16 * 512) return;
  int k, n; bsw_idx(i, 16, k, n);
  int c = k & 31, p = k >> 5, ky = p >> 2, kx = p & 3;
  d[i] = (_Float16)w2[n * 512 + c * 16 + ky * 4 + kx];
}
__global__ void k_swz3(const float* __restrict__ w3, _Float16* __restrict__ d) {
  int i = blockIdx.x * 256 + threadIdx.x; if (i >= 4 * 18 * 512) return;
  int k, n; bsw_idx(i, 18, k, n);
  int c = k & 63, p = k >> 6, ky = p / 3, kx = p % 3;
  d[i] = (_Float16)w3[n * 576 + c * 9 + ky * 3 + kx];
}
__global__ void k_swz_kn(const float* __restrict__ s, _Float16* __restrict__ d,
                         int N, int KC, int total) {
  int i = blockIdx.x * 256 + threadIdx.x; if (i >= total) return;
  int k, n; bsw_idx(i, KC, k, n);
  d[i] = (_Float16)s[k * N + n];
}
__global__ void k_swz_head(const float* __restrict__ wpo1,
                           const float* __restrict__ wv1, _Float16* __restrict__ d) {
  int i = blockIdx.x * 256 + threadIdx.x; if (i >= 8 * 256 * 512) return;
  int k, n; bsw_idx(i, 256, k, n);
  d[i] = (_Float16)((n < 64) ? wpo1[k * 64 + n] : wv1[k * 64 + (n - 64)]);
}

// ---- conv1: M=2048*400, K=192 (ky,kx,c3), N=32 ---------------------------
__global__ __launch_bounds__(256) void k_conv1(const float* __restrict__ img,
    const _Float16* __restrict__ Bsw, const float* __restrict__ bias,
    _Float16* __restrict__ a1h) {
  __shared__ _Float16 lep[8 * 512];
  const int Mt = (kTB * 400) / 16;                 // 51200 waves
  const int warp = threadIdx.x >> 5;
  const int wt = blockIdx.x * 8 + warp;
  if (wt >= Mt) return;
  const int lane = threadIdx.x & 31;
  const int m = wt * 16 + (lane & 15);
  const int tb = m / 400, r = m % 400, oy = r / 20, ox = r % 20;
  const float* src = img + ((size_t)tb * 84 * 84 + oy * 4 * 84 + ox * 4) * 3;
  const int alo = (lane >> 4) * 8;
  v8f acc0 = {}, acc1 = {};
  for (int kc = 0; kc < 6; ++kc) {
    v16h a;
#pragma unroll
    for (int h = 0; h < 2; ++h) {
      int ks = kc * 32 + alo + h * 16;             // 24%8==0: run stays in one ky row
      const float* p = src + (ks / 24) * 252 + (ks % 24);
#pragma unroll
      for (int i = 0; i < 8; ++i) a[h * 8 + i] = (_Float16)(p[i] * (1.0f / 255.0f));
    }
    v16h b0 = *(const v16h*)(Bsw + ((size_t)(kc * 32 + lane)) * 16);
    v16h b1 = *(const v16h*)(Bsw + ((size_t)((6 + kc) * 32 + lane)) * 16);
    acc0 = wmma(a, b0, acc0);
    acc1 = wmma(a, b1, acc1);
  }
  epi_store32(a1h, 32, wt * 16, 0, acc0, acc1, bias[lane & 15], bias[16 + (lane & 15)],
              true, lep + warp * 512, lane);
}

// ---- conv2: M=2048*81, K=512 (ky,kx,c32), N=64 ---------------------------
__global__ __launch_bounds__(256) void k_conv2(const _Float16* __restrict__ a1h,
    const _Float16* __restrict__ Bsw, const float* __restrict__ bias,
    _Float16* __restrict__ a2h) {
  __shared__ _Float16 lep[8 * 512];
  const int Mt = (kTB * 81) / 16;                  // 10368
  const int warp = threadIdx.x >> 5;
  const int wt = blockIdx.x * 8 + warp;
  if (wt >= Mt * 2) return;
  const int np = wt / Mt, mt = wt % Mt;
  const int lane = threadIdx.x & 31;
  const int m = mt * 16 + (lane & 15);
  const int tb = m / 81, r = m % 81, oy = r / 9, ox = r % 9;
  const int iy0 = oy * 2, ix0 = ox * 2, tbase = tb * 400;
  const int alo = (lane >> 4) * 8;
  const size_t nb0 = (size_t)(np * 2) * 16 * 512, nb1 = (size_t)(np * 2 + 1) * 16 * 512;
  v8f acc0 = {}, acc1 = {};
  for (int kc = 0; kc < 16; ++kc) {                // pixel index == kc
    int ky = kc >> 2, kx = kc & 3;
    const _Float16* ap = a1h + (size_t)(tbase + (iy0 + ky) * 20 + ix0 + kx) * 32;
    v16h a = cat(*(const v8h*)(ap + alo), *(const v8h*)(ap + alo + 16));
    v16h b0 = *(const v16h*)(Bsw + nb0 + ((size_t)(kc * 32 + lane)) * 16);
    v16h b1 = *(const v16h*)(Bsw + nb1 + ((size_t)(kc * 32 + lane)) * 16);
    acc0 = wmma(a, b0, acc0);
    acc1 = wmma(a, b1, acc1);
  }
  const int n0 = np * 32 + (lane & 15);
  epi_store32(a2h, 64, mt * 16, np * 32, acc0, acc1, bias[n0], bias[n0 + 16],
              true, lep + warp * 512, lane);
}

// ---- conv3: M=2048*49, K=576 (ky,kx,c64), N=64; out NCHW-flat [tb][3136] -
__global__ __launch_bounds__(256) void k_conv3(const _Float16* __restrict__ a2h,
    const _Float16* __restrict__ Bsw, const float* __restrict__ bias,
    _Float16* __restrict__ a3h) {
  const int Mt = (kTB * 49) / 16;                  // 6272
  const int wt = blockIdx.x * 8 + (threadIdx.x >> 5);
  if (wt >= Mt * 2) return;
  const int np = wt / Mt, mt = wt % Mt;
  const int lane = threadIdx.x & 31;
  const int m = mt * 16 + (lane & 15);
  const int tb = m / 49, r = m % 49, oy = r / 7, ox = r % 7;
  const int tbase = tb * 81;
  const int alo = (lane >> 4) * 8;
  const size_t nb0 = (size_t)(np * 2) * 18 * 512, nb1 = (size_t)(np * 2 + 1) * 18 * 512;
  v8f acc0 = {}, acc1 = {};
  for (int kc = 0; kc < 18; ++kc) {                // pixel = kc>>1, c-half = kc&1
    int p = kc >> 1, ky = p / 3, kx = p % 3;
    const _Float16* ap = a2h + (size_t)(tbase + (oy + ky) * 9 + ox + kx) * 64
                       + (kc & 1) * 32;
    v16h a = cat(*(const v8h*)(ap + alo), *(const v8h*)(ap + alo + 16));
    v16h b0 = *(const v16h*)(Bsw + nb0 + ((size_t)(kc * 32 + lane)) * 16);
    v16h b1 = *(const v16h*)(Bsw + nb1 + ((size_t)(kc * 32 + lane)) * 16);
    acc0 = wmma(a, b0, acc0);
    acc1 = wmma(a, b1, acc1);
  }
  const int hi = lane >> 4, n0 = np * 32 + (lane & 15), n1 = n0 + 16;
  const float bs0 = bias[n0], bs1 = bias[n1];
#pragma unroll
  for (int rI = 0; rI < 8; ++rI) {                 // scatter (stride-49), keep scalar
    int mr = mt * 16 + rI + 8 * hi;
    int tb2 = mr / 49, r2 = mr % 49, oy2 = r2 / 7, ox2 = r2 % 7;
    float v0 = acc0[rI] + bs0; v0 = v0 > 0.f ? v0 : 0.f;
    float v1 = acc1[rI] + bs1; v1 = v1 > 0.f ? v1 : 0.f;
    a3h[tb2 * 3136 + n0 * 49 + oy2 * 7 + ox2] = (_Float16)v0;
    a3h[tb2 * 3136 + n1 * 49 + oy2 * 7 + ox2] = (_Float16)v1;
  }
}

// ---- fc: M=2048, K=3136, N=512 -------------------------------------------
__global__ __launch_bounds__(256) void k_fc(const _Float16* __restrict__ a3h,
    const _Float16* __restrict__ Bsw, const float* __restrict__ bias,
    _Float16* __restrict__ hh) {
  __shared__ _Float16 lep[8 * 512];
  const int Mt = 128, KC = 98;
  const int warp = threadIdx.x >> 5;
  const int wt = blockIdx.x * 8 + warp;
  if (wt >= Mt * 16) return;
  const int np = wt / Mt, mt = wt % Mt;
  const int lane = threadIdx.x & 31;
  const int m = mt * 16 + (lane & 15);
  const _Float16* ap = a3h + (size_t)m * 3136;
  const int alo = (lane >> 4) * 8;
  const size_t nb0 = (size_t)(np * 2) * KC * 512, nb1 = (size_t)(np * 2 + 1) * KC * 512;
  v8f acc0 = {}, acc1 = {};
  for (int kc = 0; kc < KC; ++kc) {
    __builtin_prefetch(ap + (kc + 2) * 32, 0, 0);
    v16h a = cat(*(const v8h*)(ap + kc * 32 + alo),
                 *(const v8h*)(ap + kc * 32 + alo + 16));
    v16h b0 = *(const v16h*)(Bsw + nb0 + ((size_t)(kc * 32 + lane)) * 16);
    v16h b1 = *(const v16h*)(Bsw + nb1 + ((size_t)(kc * 32 + lane)) * 16);
    acc0 = wmma(a, b0, acc0);
    acc1 = wmma(a, b1, acc1);
  }
  const int n0 = np * 32 + (lane & 15);
  epi_store32(hh, 512, mt * 16, np * 32, acc0, acc1, bias[n0], bias[n0 + 16],
              true, lep + warp * 512, lane);
}

// ---- Ww: M=2048, K=512, N=32 -> wfeat f32 (+bias, no relu) ---------------
__global__ __launch_bounds__(256) void k_ww(const _Float16* __restrict__ hh,
    const _Float16* __restrict__ Bsw, const float* __restrict__ bias,
    float* __restrict__ wfeat) {
  const int Mt = 128, KC = 16;
  const int wt = blockIdx.x * 8 + (threadIdx.x >> 5);
  if (wt >= Mt) return;
  const int lane = threadIdx.x & 31;
  const int m = wt * 16 + (lane & 15);
  const _Float16* ap = hh + (size_t)m * 512;
  const int alo = (lane >> 4) * 8;
  v8f acc0 = {}, acc1 = {};
  for (int kc = 0; kc < KC; ++kc) {
    v16h a = cat(*(const v8h*)(ap + kc * 32 + alo),
                 *(const v8h*)(ap + kc * 32 + alo + 16));
    v16h b0 = *(const v16h*)(Bsw + ((size_t)(kc * 32 + lane)) * 16);
    v16h b1 = *(const v16h*)(Bsw + ((size_t)((KC + kc) * 32 + lane)) * 16);
    acc0 = wmma(a, b0, acc0);
    acc1 = wmma(a, b1, acc1);
  }
  const int hi = lane >> 4, n0 = lane & 15, n1 = n0 + 16;
  const float bs0 = bias[n0], bs1 = bias[n1];
#pragma unroll
  for (int rI = 0; rI < 8; ++rI) {
    int mr = wt * 16 + rI + 8 * hi;
    wfeat[mr * 32 + n0] = acc0[rI] + bs0;
    wfeat[mr * 32 + n1] = acc1[rI] + bs1;
  }
}

// ---- recurrent map scan: thread carries 4 state elems (x-quad) over T ----
__global__ __launch_bounds__(256) void k_scan(const float* __restrict__ state0,
    const float* __restrict__ done, const int* __restrict__ pos,
    const float* __restrict__ wfeat, _Float16* __restrict__ hidh,
    float* __restrict__ out_state) {
  int gid = blockIdx.x * 256 + threadIdx.x;        // < 64*2048
  int b = gid >> 11, q = gid & 2047;
  int e0 = q * 4;                                  // f*256 + y*16 + x0
  int f = e0 >> 8, y = (e0 >> 4) & 15, x0 = e0 & 15;
  v4f s = *(const v4f*)(state0 + (size_t)b * kMAP + e0);
  for (int t = 0; t < kT; ++t) {
    int idx = t * kB + b;
    float keep = 1.0f - done[idx];
    s *= keep;
    int p0 = pos[idx * 2], p1 = pos[idx * 2 + 1];
    if (p0 == y && (p1 - x0) >= 0 && (p1 - x0) < 4)
      s[p1 - x0] += wfeat[idx * 32 + f];
    v4h hv;
#pragma unroll
    for (int i = 0; i < 4; ++i) hv[i] = (_Float16)s[i];
    *(v4h*)(hidh + (size_t)idx * kMAP + e0) = hv;
  }
  *(v4f*)(out_state + (size_t)b * kMAP + e0) = s;
}

// ---- head GEMM: hidden[2048,8192] @ [wpo1|wv1][8192,128] -----------------
// B staged to LDS 4 chunks at a time (async-to-LDS when available).
__global__ __launch_bounds__(256) void k_head(const _Float16* __restrict__ hidh,
    const _Float16* __restrict__ Bsw, float* __restrict__ head) {
  __shared__ _Float16 lb[4096];                    // 4 chunks x 2 tiles x 512 halfs
  const int KC = 256;
  const int warp = threadIdx.x >> 5, lane = threadIdx.x & 31;
  const int mtg = blockIdx.x & 15;                 // 16 m-groups of 8 waves
  const int np  = blockIdx.x >> 4;                 // 4 n-pairs
  const int mt = mtg * 8 + warp;
  const int m = mt * 16 + (lane & 15);
  const _Float16* ap = hidh + (size_t)m * kMAP;
  const int alo = (lane >> 4) * 8;
  const int cq = warp >> 1, tile = warp & 1;       // staging role of this wave
  const _Float16* gsrc = Bsw + (size_t)(np * 2 + tile) * KC * 512 + (size_t)lane * 16;
  _Float16* ldst = lb + cq * 1024 + tile * 512 + lane * 16;
  v8f acc0 = {}, acc1 = {};
  for (int kb = 0; kb < KC / 4; ++kb) {
    __syncthreads();
    const _Float16* g = gsrc + (size_t)(kb * 4 + cq) * 512;
    cp16_to_lds(g, ldst);
    cp16_to_lds(g + 8, ldst + 8);
    async_wait();
    __syncthreads();
    __builtin_prefetch(ap + (kb * 4 + 8) * 32, 0, 0);
#pragma unroll
    for (int c = 0; c < 4; ++c) {
      int kc = kb * 4 + c;
      v16h a = cat(*(const v8h*)(ap + kc * 32 + alo),
                   *(const v8h*)(ap + kc * 32 + alo + 16));
      v16h b0 = *(const v16h*)(lb + c * 1024 + lane * 16);
      v16h b1 = *(const v16h*)(lb + c * 1024 + 512 + lane * 16);
      acc0 = wmma(a, b0, acc0);
      acc1 = wmma(a, b1, acc1);
    }
  }
  const int hi = lane >> 4, n0 = np * 32 + (lane & 15), n1 = n0 + 16;
#pragma unroll
  for (int rI = 0; rI < 8; ++rI) {
    int mr = mt * 16 + rI + 8 * hi;
    head[mr * 128 + n0] = acc0[rI];
    head[mr * 128 + n1] = acc1[rI];
  }
}

// ---- position MLP: one-hot input -> row gathers of wp1 -------------------
__global__ __launch_bounds__(64) void k_pos(const int* __restrict__ pos,
    const float* __restrict__ wp1, const float* __restrict__ bp1,
    const float* __restrict__ wp2, const float* __restrict__ bp2,
    float* __restrict__ posf) {
  __shared__ float z[64];
  int tb = blockIdx.x, i = threadIdx.x;
  int p0 = pos[tb * 2], p1 = pos[tb * 2 + 1];
  float zz = wp1[p0 * 64 + i] + wp1[(16 + p1) * 64 + i] + bp1[i];
  z[i] = zz > 0.f ? zz : 0.f;
  __syncthreads();
  float s = bp2[i];
  for (int k = 0; k < 64; ++k) s += z[k] * wp2[k * 64 + i];
  posf[tb * 64 + i] = s;
}

// ---- tail: add pos-feature slice, relu, tiny output layers ---------------
__global__ __launch_bounds__(64) void k_tail(const float* __restrict__ head,
    const float* __restrict__ posf,
    const float* __restrict__ wpo1, const float* __restrict__ bpo1,
    const float* __restrict__ wpo2, const float* __restrict__ bpo2,
    const float* __restrict__ wv1,  const float* __restrict__ bv1,
    const float* __restrict__ wv2,  const float* __restrict__ bv2,
    float* __restrict__ out) {
  __shared__ float pf[64], ap[64], av[64];
  int tb = blockIdx.x, n = threadIdx.x;
  pf[n] = posf[tb * 64 + n];
  __syncthreads();
  float sp = head[tb * 128 + n] + bpo1[n];
  float sv = head[tb * 128 + 64 + n] + bv1[n];
  for (int i = 0; i < 64; ++i) {
    sp += pf[i] * wpo1[(kMAP + i) * 64 + n];
    sv += pf[i] * wv1[(kMAP + i) * 64 + n];
  }
  ap[n] = sp > 0.f ? sp : 0.f;
  av[n] = sv > 0.f ? sv : 0.f;
  __syncthreads();
  if (n < kNACT) {
    float s = bpo2[n];
    for (int i = 0; i < 64; ++i) s += ap[i] * wpo2[i * kNACT + n];
    out[tb * kNACT + n] = s;
  } else if (n == 8) {
    float s = bv2[0];
    for (int i = 0; i < 64; ++i) s += av[i] * wv2[i];
    out[kTB * kNACT + tb] = s;
  }
}

extern "C" void kernel_launch(void* const* d_in, const int* in_sizes, int n_in,
                              void* d_out, int out_size, void* d_ws, size_t ws_size,
                              hipStream_t stream) {
  const float* image  = (const float*)d_in[0];
  const float* done   = (const float*)d_in[1];
  const float* state0 = (const float*)d_in[2];
  const int*   pos    = (const int*)d_in[3];
  const float* w1  = (const float*)d_in[4];  const float* b1  = (const float*)d_in[5];
  const float* w2  = (const float*)d_in[6];  const float* b2  = (const float*)d_in[7];
  const float* w3  = (const float*)d_in[8];  const float* b3  = (const float*)d_in[9];
  const float* wfc = (const float*)d_in[10]; const float* bfc = (const float*)d_in[11];
  const float* Ww  = (const float*)d_in[12]; const float* bw  = (const float*)d_in[13];
  const float* wp1 = (const float*)d_in[14]; const float* bp1 = (const float*)d_in[15];
  const float* wp2 = (const float*)d_in[16]; const float* bp2 = (const float*)d_in[17];
  const float* wpo1= (const float*)d_in[18]; const float* bpo1= (const float*)d_in[19];
  const float* wpo2= (const float*)d_in[20]; const float* bpo2= (const float*)d_in[21];
  const float* wv1 = (const float*)d_in[22]; const float* bv1 = (const float*)d_in[23];
  const float* wv2 = (const float*)d_in[24]; const float* bv2 = (const float*)d_in[25];
  float* out = (float*)d_out;

  char* w = (char*)d_ws;
  auto alloc = [&](size_t bytes) {
    char* p = w; w += (bytes + 255) & ~(size_t)255; return p;
  };
  _Float16* a1h  = (_Float16*)alloc((size_t)kTB * 400 * 32 * 2);
  _Float16* a2h  = (_Float16*)alloc((size_t)kTB * 81 * 64 * 2);
  _Float16* a3h  = (_Float16*)alloc((size_t)kTB * 3136 * 2);
  _Float16* hh   = (_Float16*)alloc((size_t)kTB * 512 * 2);
  float*    wfeat= (float*)   alloc((size_t)kTB * 32 * 4);
  _Float16* hidh = (_Float16*)alloc((size_t)kTB * kMAP * 2);
  float*    posf = (float*)   alloc((size_t)kTB * 64 * 4);
  float*    headb= (float*)   alloc((size_t)kTB * 128 * 4);
  _Float16* b1s  = (_Float16*)alloc((size_t)2 * 6 * 512 * 2);
  _Float16* b2s  = (_Float16*)alloc((size_t)4 * 16 * 512 * 2);
  _Float16* b3s  = (_Float16*)alloc((size_t)4 * 18 * 512 * 2);
  _Float16* wfcs = (_Float16*)alloc((size_t)32 * 98 * 512 * 2);
  _Float16* wws  = (_Float16*)alloc((size_t)2 * 16 * 512 * 2);
  _Float16* bhds = (_Float16*)alloc((size_t)8 * 256 * 512 * 2);

  // weight prep (fragment-swizzled f16 B matrices)
  k_swz1<<<(2 * 6 * 512 + 255) / 256, 256, 0, stream>>>(w1, b1s);
  k_swz2<<<(4 * 16 * 512 + 255) / 256, 256, 0, stream>>>(w2, b2s);
  k_swz3<<<(4 * 18 * 512 + 255) / 256, 256, 0, stream>>>(w3, b3s);
  k_swz_kn<<<(32 * 98 * 512 + 255) / 256, 256, 0, stream>>>(wfc, wfcs, 512, 98,
                                                            32 * 98 * 512);
  k_swz_kn<<<(2 * 16 * 512 + 255) / 256, 256, 0, stream>>>(Ww, wws, 32, 16,
                                                           2 * 16 * 512);
  k_swz_head<<<(8 * 256 * 512 + 255) / 256, 256, 0, stream>>>(wpo1, wv1, bhds);

  // trunk
  k_conv1<<<51200 / 8, 256, 0, stream>>>(image, b1s, b1, a1h);
  k_conv2<<<(10368 * 2) / 8, 256, 0, stream>>>(a1h, b2s, b2, a2h);
  k_conv3<<<(6272 * 2) / 8, 256, 0, stream>>>(a2h, b3s, b3, a3h);
  k_fc<<<(128 * 16) / 8, 256, 0, stream>>>(a3h, wfcs, bfc, hh);
  k_ww<<<128 / 8, 256, 0, stream>>>(hh, wws, bw, wfeat);

  // recurrent map + readout
  k_scan<<<(kB * 2048) / 256, 256, 0, stream>>>(state0, done, pos, wfeat, hidh,
                                                out + kTB * kNACT + kTB);
  k_head<<<64, 256, 0, stream>>>(hidh, bhds, headb);
  k_pos<<<kTB, 64, 0, stream>>>(pos, wp1, bp1, wp2, bp2, posf);
  k_tail<<<kTB, 64, 0, stream>>>(headb, posf, wpo1, bpo1, wpo2, bpo2,
                                 wv1, bv1, wv2, bv2, out);
}
